// SJN_Meta_23673859735573
// MI455X (gfx1250) — matrix operand order repaired and checked by reading, not compiled, not run on hardware
//
#include <hip/hip_runtime.h>
#include <hip/hip_bf16.h>
#include <math.h>

// ---------------------------------------------------------------------------
// CDNA5 / gfx1250 implementation of the 3-MetaLayer GraphNet forward pass.
// All dense math runs through v_wmma_f32_16x16x32_f16 (wave32 WMMA).
// BatchNorm (eval mode) is folded into f16 weights on-device; epilogue is
// bias + ReLU + f16 store. Scatter-mean uses global_atomic_add_f32.
// The global-MLP branch is dead code w.r.t. the outputs and is skipped.
//
// GEMM: each wave computes 32 rows x (NCT*16) cols (2 row tiles reuse every
// B fragment; NCT col tiles reuse every A fragment), with next-K A fragments
// prefetched ahead of the WMMA chain so the scheduler can overlap loads with
// matrix math instead of draining loadcnt to zero per WMMA.
// ---------------------------------------------------------------------------

typedef __attribute__((ext_vector_type(16))) _Float16 v16h;
typedef __attribute__((ext_vector_type(8)))  _Float16 v8h;
typedef __attribute__((ext_vector_type(8)))  float    v8f;

#define NNODES 200000
#define NEDGES 1000000

// ----------------------------- utility kernels -----------------------------

__global__ void k_zero_f32(float* p, long long n) {
    long long i = (long long)blockIdx.x * blockDim.x + threadIdx.x;
    if (i < n) p[i] = 0.f;
}

__global__ void k_zero_i32(int* p, int n) {
    int i = blockIdx.x * blockDim.x + threadIdx.x;
    if (i < n) p[i] = 0;
}

__global__ void k_count(const int* __restrict__ col, int* __restrict__ cnt, int E) {
    int e = blockIdx.x * blockDim.x + threadIdx.x;
    if (e < E) atomicAdd(&cnt[col[e]], 1);
}

__global__ void k_cvt_x(const float* __restrict__ x, _Float16* __restrict__ xh, int n) {
    int i = blockIdx.x * blockDim.x + threadIdx.x;
    if (i < n) xh[i] = (_Float16)x[i];
}

// ea fp32 [E,19] -> f16 [E,32] zero-padded
__global__ void k_cvt_ea(const float* __restrict__ ea, _Float16* __restrict__ eah, int E) {
    int idx = blockIdx.x * blockDim.x + threadIdx.x;
    int e = idx >> 5, k = idx & 31;
    if (e >= E) return;
    eah[idx] = (k < 19) ? (_Float16)ea[(size_t)e * 19 + k] : (_Float16)0.f;
}

// Fold BN (eval) into linear: W' = W * s, b' = (b-m)*s + beta, s = g*rsqrt(v+eps).
// Writes zero-padded f16 weights [Opad,Kpad] and f32 bias [Opad].
__global__ void k_fold(const float* __restrict__ W, const float* __restrict__ b,
                       const float* __restrict__ g, const float* __restrict__ beta,
                       const float* __restrict__ mm, const float* __restrict__ vv,
                       _Float16* __restrict__ Wf, float* __restrict__ bf,
                       int O, int K, int Opad, int Kpad, int has_bn) {
    long long idx = (long long)blockIdx.x * blockDim.x + threadIdx.x;
    if (idx >= (long long)Opad * Kpad) return;
    int o = (int)(idx / Kpad), k = (int)(idx % Kpad);
    float s = 1.f, bb = 0.f;
    if (o < O) {
        if (has_bn) {
            float sc = g[o] * rsqrtf(vv[o] + 1e-5f);
            s = sc;
            bb = (b[o] - mm[o]) * sc + beta[o];
        } else {
            bb = b[o];
        }
    }
    Wf[idx] = (o < O && k < K) ? (_Float16)(W[(long long)o * K + k] * s) : (_Float16)0.f;
    if (k == 0) bf[o] = bb;
}

// cat([x[row], x[col], ea]) -> [E,64] f16 (51 real cols, rest zero)
__global__ void k_pack_edge(const _Float16* __restrict__ xh, const _Float16* __restrict__ eah,
                            const int* __restrict__ row, const int* __restrict__ col,
                            _Float16* __restrict__ A, int E) {
    int idx = blockIdx.x * blockDim.x + threadIdx.x;
    int e = idx >> 6, k = idx & 63;
    if (e >= E) return;
    _Float16 v = (_Float16)0.f;
    if (k < 16)       v = xh[(size_t)row[e] * 16 + k];
    else if (k < 32)  v = xh[(size_t)col[e] * 16 + (k - 16)];
    else if (k < 51)  v = eah[(size_t)e * 32 + (k - 32)];
    A[(size_t)e * 64 + k] = v;
}

// cat([x[row], ea]) -> [E,64] f16 (35 real cols)
__global__ void k_pack_msg(const _Float16* __restrict__ xh, const _Float16* __restrict__ eah,
                           const int* __restrict__ row, _Float16* __restrict__ A, int E) {
    int idx = blockIdx.x * blockDim.x + threadIdx.x;
    int e = idx >> 6, k = idx & 63;
    if (e >= E) return;
    _Float16 v = (_Float16)0.f;
    if (k < 16)       v = xh[(size_t)row[e] * 16 + k];
    else if (k < 35)  v = eah[(size_t)e * 32 + (k - 16)];
    A[(size_t)e * 64 + k] = v;
}

// cat([x, agg_sum/count]) -> [N,Kp] f16
__global__ void k_pack_node(const _Float16* __restrict__ xh, const float* __restrict__ agg,
                            const int* __restrict__ cnt, _Float16* __restrict__ A,
                            int N, int ls, int Kp) {
    long long idx = (long long)blockIdx.x * blockDim.x + threadIdx.x;
    if (idx >= (long long)N * Kp) return;
    int n = (int)(idx / Kp), k = (int)(idx % Kp);
    _Float16 v = (_Float16)0.f;
    if (k < 16) {
        v = xh[(size_t)n * 16 + k];
    } else if (k < 16 + ls) {
        float c = fmaxf((float)cnt[n], 1.f);
        v = (_Float16)(agg[(size_t)n * ls + (k - 16)] / c);
    }
    A[idx] = v;
}

// scatter-add of edge messages M [E,ls] f16 into agg [N,ls] f32 keyed by col
__global__ void k_seg_sum(const _Float16* __restrict__ M, const int* __restrict__ col,
                          float* __restrict__ agg, int E, int ls) {
    long long idx = (long long)blockIdx.x * blockDim.x + threadIdx.x;
    if (idx >= (long long)E * ls) return;
    int e = (int)(idx / ls), j = (int)(idx % ls);
    atomicAdd(&agg[(size_t)col[e] * ls + j], (float)M[idx]);
}

// ------------------------------ WMMA GEMM ----------------------------------
// Y[R,Opad] = relu(A[R,K] * W[Opad,K]^T + bias), f16 in/out, fp32 accumulate.
// One wave -> 32 rows (2 row tiles) x (NCT*16) cols; 2 waves per block.
// R % 64 == 0, K % 32 == 0, Opad % (NCT*16) == 0 guaranteed by caller,
// so EXEC is all-ones at every WMMA (ISA requirement).
template <int NCT>
__global__ void __launch_bounds__(64)
k_gemm(const _Float16* __restrict__ A, const _Float16* __restrict__ W,
       const float* __restrict__ bias, _Float16* __restrict__ Y,
       int R, int K, int Opad) {
    const int lane = threadIdx.x & 31;
    const int wave = threadIdx.x >> 5;
    const int rowTile = (blockIdx.x * 2 + wave) * 2;   // two consecutive 16-row tiles
    const int colBase = blockIdx.y * (NCT * 16);
    const int li = lane & 15;
    const int kh = lane >> 4;               // lane half selects K sub-blocks

    const _Float16* arow0 = A + (size_t)(rowTile * 16 + li) * K;
    const _Float16* arow1 = arow0 + (size_t)16 * K;

    v8f acc0[NCT], acc1[NCT];
#pragma unroll
    for (int t = 0; t < NCT; ++t) {
        v8f z = {0.f, 0.f, 0.f, 0.f, 0.f, 0.f, 0.f, 0.f};
        acc0[t] = z;
        acc1[t] = z;
    }

    // A fragment layout (ISA 16-bit A): lane(li)=row M; halves 0..7 =
    // K[kh*8..+7], halves 8..15 = K[16+kh*8..+7]. Preload K-step 0.
    v16h a0, a1;
    ((v8h*)&a0)[0] = *(const v8h*)(arow0 + kh * 8);
    ((v8h*)&a0)[1] = *(const v8h*)(arow0 + 16 + kh * 8);
    ((v8h*)&a1)[0] = *(const v8h*)(arow1 + kh * 8);
    ((v8h*)&a1)[1] = *(const v8h*)(arow1 + 16 + kh * 8);

    for (int k0 = 0; k0 < K; k0 += 32) {
        // B fragments for this K-step, all issued up front (clause-able).
        // B lane(li)=col N; halves 0..7 = K[kh*8..+7], 8..15 = K[16+kh*8..+7]. Column N of
        // B == row N of W (row-major [Opad,K]) -> contiguous 16B loads.
        v16h b[NCT];
#pragma unroll
        for (int t = 0; t < NCT; ++t) {
            const _Float16* wrow = W + (size_t)(colBase + t * 16 + li) * K + k0;
            ((v8h*)&b[t])[0] = *(const v8h*)(wrow + kh * 8);
            ((v8h*)&b[t])[1] = *(const v8h*)(wrow + 16 + kh * 8);
        }
        // Prefetch next K-step A fragments before the WMMA chain so the
        // loads overlap matrix math (partial s_wait_loadcnt instead of 0).
        v16h a0n = a0, a1n = a1;
        if (k0 + 32 < K) {
            const int kn = k0 + 32;
            ((v8h*)&a0n)[0] = *(const v8h*)(arow0 + kn + kh * 8);
            ((v8h*)&a0n)[1] = *(const v8h*)(arow0 + kn + 16 + kh * 8);
            ((v8h*)&a1n)[0] = *(const v8h*)(arow1 + kn + kh * 8);
            ((v8h*)&a1n)[1] = *(const v8h*)(arow1 + kn + 16 + kh * 8);
        }
#pragma unroll
        for (int t = 0; t < NCT; ++t) {
            acc0[t] = __builtin_amdgcn_wmma_f32_16x16x32_f16(
                false, a0, false, b[t], (short)0, acc0[t], false, false);
            acc1[t] = __builtin_amdgcn_wmma_f32_16x16x32_f16(
                false, a1, false, b[t], (short)0, acc1[t], false, false);
        }
        a0 = a0n;
        a1 = a1n;
    }

    // C/D layout: lanes 0-15 -> M=r, N=lane; lanes 16-31 -> M=8+r, N=lane-16
    const int mBase0 = rowTile * 16 + (kh ? 8 : 0);
    const int mBase1 = mBase0 + 16;
#pragma unroll
    for (int t = 0; t < NCT; ++t) {
        const int n = colBase + t * 16 + li;
        const float bv = bias[n];
#pragma unroll
        for (int r = 0; r < 8; ++r) {
            float v0 = fmaxf(acc0[t][r] + bv, 0.f);
            float v1 = fmaxf(acc1[t][r] + bv, 0.f);
            Y[(size_t)(mBase0 + r) * Opad + n] = (_Float16)v0;
            Y[(size_t)(mBase1 + r) * Opad + n] = (_Float16)v1;
        }
    }
}

// ------------------------------ output heads -------------------------------

__global__ void k_head_node(const _Float16* __restrict__ xh, const float* __restrict__ Wl,
                            const float* __restrict__ bl, float* __restrict__ out, int N) {
    int n = blockIdx.x * blockDim.x + threadIdx.x;
    if (n >= N) return;
    float s = bl[0];
#pragma unroll
    for (int k = 0; k < 16; ++k) s += (float)xh[(size_t)n * 16 + k] * Wl[k];
    out[n] = 1.f / (1.f + expf(-s));
}

__global__ void k_head_edge(const _Float16* __restrict__ eah, const float* __restrict__ Wl,
                            const float* __restrict__ bl, float* __restrict__ out, int E) {
    int e = blockIdx.x * blockDim.x + threadIdx.x;
    if (e >= E) return;
    float s = bl[0];
#pragma unroll
    for (int k = 0; k < 19; ++k) s += (float)eah[(size_t)e * 32 + k] * Wl[k];
    out[e] = 1.f / (1.f + expf(-s));
}

// ------------------------------- host side ---------------------------------

static inline size_t align256(size_t x) { return (x + 255) & ~(size_t)255; }

static void launch_gemm(const _Float16* A, const _Float16* W, const float* b, _Float16* Y,
                        int R, int K, int Opad, hipStream_t s) {
    dim3 blk(64);                       // 2 waves, each 32 rows -> 64 rows/block
    int tiles = Opad / 16;
    if (tiles % 4 == 0) {
        dim3 g(R / 64, tiles / 4);
        k_gemm<4><<<g, blk, 0, s>>>(A, W, b, Y, R, K, Opad);
    } else if (tiles % 2 == 0) {
        dim3 g(R / 64, tiles / 2);
        k_gemm<2><<<g, blk, 0, s>>>(A, W, b, Y, R, K, Opad);
    } else {
        dim3 g(R / 64, tiles);
        k_gemm<1><<<g, blk, 0, s>>>(A, W, b, Y, R, K, Opad);
    }
}

static void launch_fold(const float* W, const float* b, const float* g, const float* beta,
                        const float* mm, const float* vv, _Float16* Wf, float* bf,
                        int O, int K, int Opad, int Kpad, int has_bn, hipStream_t s) {
    long long tot = (long long)Opad * Kpad;
    int grid = (int)((tot + 255) / 256);
    k_fold<<<grid, 256, 0, s>>>(W, b, g, beta, mm, vv, Wf, bf, O, K, Opad, Kpad, has_bn);
}

extern "C" void kernel_launch(void* const* d_in, const int* in_sizes, int n_in,
                              void* d_out, int out_size, void* d_ws, size_t ws_size,
                              hipStream_t stream) {
    (void)in_sizes; (void)n_in; (void)out_size; (void)ws_size;
    const int N = NNODES, E = NEDGES;

    const float* x0   = (const float*)d_in[0];
    const int*   eidx = (const int*)d_in[1];
    const float* ea0  = (const float*)d_in[2];
    // d_in[3]=u, d_in[4]=batch: unused (global branch does not affect outputs)
    const int* row = eidx;
    const int* col = eidx + E;

    // --- workspace bump allocator (with lifetime-based aliasing) ---
    char* ws = (char*)d_ws;
    size_t off = 0;
    auto take = [&](size_t bytes) -> char* { char* p = ws + off; off += align256(bytes); return p; };
    _Float16* xA   = (_Float16*)take((size_t)N * 16 * 2);
    _Float16* xB   = (_Float16*)take((size_t)N * 16 * 2);
    _Float16* eaH  = (_Float16*)take((size_t)E * 32 * 2);
    _Float16* Abuf = (_Float16*)take((size_t)E * 64 * 2);
    _Float16* H1   = (_Float16*)take((size_t)E * 256 * 2);   // edge hidden 1 / messages
    _Float16* H2   = (_Float16*)take((size_t)E * 256 * 2);   // edge hidden 2
    int*      cnt  = (int*)take((size_t)N * 4);
    char*     arena = take((size_t)2 << 20);                 // folded weights
    // aliases (lifetimes disjoint on the in-order stream):
    float*    agg = (float*)H2;                              // [N,256] f32, after H2 consumed
    _Float16* NP  = (_Float16*)((char*)H2 + ((size_t)224 << 20)); // node packed [N,<=288]
    _Float16* NH  = (_Float16*)H1;                           // node hidden, after H1 consumed

    // --- one-time conversions & static graph counts ---
    k_cvt_x<<<(N * 16 + 255) / 256, 256, 0, stream>>>(x0, xA, N * 16);
    k_cvt_ea<<<(E * 32 + 255) / 256, 256, 0, stream>>>(ea0, eaH, E);
    k_zero_i32<<<(N + 255) / 256, 256, 0, stream>>>(cnt, N);
    k_count<<<(E + 255) / 256, 256, 0, stream>>>(col, cnt, E);

    _Float16* curx = xA;
    _Float16* nxtx = xB;
    const int LS[3] = {64, 256, 128};

    for (int m = 0; m < 3; ++m) {
        const int ls = LS[m];
        const int se = 16 + ls;
        const int Kp = (se + 31) & ~31;     // 96 / 288 / 160

        // param leaf index: 5 + ml*42 + block*14 + item
        // items: 0 W1,1 b1,2 g1,3 beta1,4 m1,5 v1,6 W2,7 b2,8 g2,9 beta2,10 m2,11 v2,12 Wo,13 bo
        auto PP = [&](int blk, int item) -> const float* {
            return (const float*)d_in[5 + m * 42 + blk * 14 + item];
        };

        // --- fold weights into arena (f16, padded) ---
        size_t ao = 0;
        auto atake = [&](size_t b) -> char* { char* p = arena + ao; ao += align256(b); return p; };
        _Float16* WfE1 = (_Float16*)atake((size_t)ls * 64 * 2);  float* bE1 = (float*)atake(ls * 4);
        _Float16* WfE2 = (_Float16*)atake((size_t)ls * ls * 2);  float* bE2 = (float*)atake(ls * 4);
        _Float16* WfE3 = (_Float16*)atake((size_t)32 * ls * 2);  float* bE3 = (float*)atake(32 * 4);
        _Float16* WfN1 = (_Float16*)atake((size_t)ls * 64 * 2);  float* bN1 = (float*)atake(ls * 4);
        _Float16* WfN2 = (_Float16*)atake((size_t)Kp * Kp * 2);  float* bN2 = (float*)atake(Kp * 4);
        _Float16* WfN3 = (_Float16*)atake((size_t)16 * Kp * 2);  float* bN3 = (float*)atake(16 * 4);

        launch_fold(PP(0,0), PP(0,1), PP(0,2), PP(0,3), PP(0,4), PP(0,5),
                    WfE1, bE1, ls, 51, ls, 64, 1, stream);
        launch_fold(PP(0,6), PP(0,7), PP(0,8), PP(0,9), PP(0,10), PP(0,11),
                    WfE2, bE2, ls, ls, ls, ls, 1, stream);
        launch_fold(PP(0,12), PP(0,13), nullptr, nullptr, nullptr, nullptr,
                    WfE3, bE3, 19, ls, 32, ls, 0, stream);
        launch_fold(PP(1,0), PP(1,1), PP(1,2), PP(1,3), PP(1,4), PP(1,5),
                    WfN1, bN1, ls, 35, ls, 64, 1, stream);
        launch_fold(PP(1,6), PP(1,7), PP(1,8), PP(1,9), PP(1,10), PP(1,11),
                    WfN2, bN2, se, se, Kp, Kp, 1, stream);
        launch_fold(PP(1,12), PP(1,13), nullptr, nullptr, nullptr, nullptr,
                    WfN3, bN3, 16, se, 16, Kp, 0, stream);

        // --- edge update ---
        k_pack_edge<<<(E * 64 + 255) / 256, 256, 0, stream>>>(curx, eaH, row, col, Abuf, E);
        launch_gemm(Abuf, WfE1, bE1, H1, E, 64, ls, stream);
        launch_gemm(H1,   WfE2, bE2, H2, E, ls, ls, stream);
        launch_gemm(H2,   WfE3, bE3, eaH, E, ls, 32, stream);   // new ea (relu), padded to 32

        // --- node update ---
        k_pack_msg<<<(E * 64 + 255) / 256, 256, 0, stream>>>(curx, eaH, row, Abuf, E);
        launch_gemm(Abuf, WfN1, bN1, H1, E, 64, ls, stream);    // messages M [E,ls]
        {
            long long na = (long long)N * ls;
            k_zero_f32<<<(int)((na + 255) / 256), 256, 0, stream>>>(agg, na);
            long long ea_tot = (long long)E * ls;
            k_seg_sum<<<(int)((ea_tot + 255) / 256), 256, 0, stream>>>(H1, col, agg, E, ls);
            long long np_tot = (long long)N * Kp;
            k_pack_node<<<(int)((np_tot + 255) / 256), 256, 0, stream>>>(curx, agg, cnt, NP, N, ls, Kp);
        }
        launch_gemm(NP, WfN2, bN2, NH, N, Kp, Kp, stream);
        launch_gemm(NH, WfN3, bN3, nxtx, N, Kp, 16, stream);    // new x [N,16] (relu)

        _Float16* t = curx; curx = nxtx; nxtx = t;
    }

    // --- output heads: sigmoid(linear) ---
    const float* Wx = (const float*)d_in[131];
    const float* bx = (const float*)d_in[132];
    const float* We = (const float*)d_in[133];
    const float* be = (const float*)d_in[134];
    float* out = (float*)d_out;
    k_head_node<<<(N + 255) / 256, 256, 0, stream>>>(curx, Wx, bx, out, N);
    k_head_edge<<<(E + 255) / 256, 256, 0, stream>>>(eaH, We, be, out + N, E);
}
